// TextEncoder_47442208751795
// MI455X (gfx1250) — matrix operand order repaired and checked by reading, not compiled, not run on hardware
//
#include <hip/hip_runtime.h>
#include <hip/hip_bf16.h>

// ---------------------------------------------------------------------------
// TextEncoder on MI455X (gfx1250): every matmul-shaped stage goes through
// v_wmma_f32_16x16x32_bf16 (wave32 WMMA, f32 accumulate), with double-buffered
// LDS staging and gfx1250 async global->LDS copies for the weight tiles.
// ---------------------------------------------------------------------------

#define BATCH 2048

typedef __bf16 bf16_t;
typedef __attribute__((ext_vector_type(4)))  int    v4i;
typedef __attribute__((ext_vector_type(8)))  bf16_t v8bf;
typedef __attribute__((ext_vector_type(16))) bf16_t v16bf;
typedef __attribute__((ext_vector_type(8)))  float  v8f;

// GEMM tiling: 256 threads = 8 wave32s arranged 4 (M) x 2 (N).
// Macro tile 128 (M) x 64 (N), K step 32. Each wave: 32x32 = 2x2 WMMA tiles.
#define TM   128
#define TN   64
#define TK   32
#define ASTR 40   // LDS row stride (bf16 elems) for A tile [TM][TK]  (80B, 16B aligned)
#define BSTR 40   // LDS row stride (bf16 elems) for B tile [TN][TK]  (n-major, k contig)
#define BLOCK 256

#ifndef __has_builtin
#define __has_builtin(x) 0
#endif
#if __has_builtin(__builtin_amdgcn_global_load_async_to_lds_b128) && \
    __has_builtin(__builtin_amdgcn_s_wait_asynccnt)
#define HAVE_ASYNC_LDS 1
#else
#define HAVE_ASYNC_LDS 0
#endif

#define AS1G __attribute__((address_space(1)))
#define AS3L __attribute__((address_space(3)))

__device__ inline void async_wait_all() {
#if HAVE_ASYNC_LDS
    __builtin_amdgcn_s_wait_asynccnt(0);
#endif
}

// ---------------- WMMA fragment loads from LDS -----------------------------
// A (16x32 bf16): lane L: m = L&15, khalf=(L>>4)*8; element i covers
//   k = ((i>>3)*16) + khalf + (i&7)  -> two contiguous 8-elem (16B) chunks.
__device__ inline v16bf load_a_frag(const bf16_t* a_lds, int m_sub, int lane) {
    int m  = m_sub + (lane & 15);
    int kh = (lane >> 4) * 8;
    v8bf lo = *(const v8bf*)&a_lds[m * ASTR + kh];
    v8bf hi = *(const v8bf*)&a_lds[m * ASTR + kh + 16];
    return __builtin_shufflevector(lo, hi, 0,1,2,3,4,5,6,7,8,9,10,11,12,13,14,15);
}
// B (32x16 bf16): lane L: n = L&15, kgroup=(L>>4)*8; element i covers
//   k = ((i>>3)*16) + kgroup + (i&7)  -> two contiguous 8-elem (16B) chunks.
__device__ inline v16bf load_b_frag(const bf16_t* b_lds, int n_sub, int lane) {
    int n  = n_sub + (lane & 15);
    int kg = (lane >> 4) * 8;
    v8bf lo = *(const v8bf*)&b_lds[n * BSTR + kg];
    v8bf hi = *(const v8bf*)&b_lds[n * BSTR + kg + 16];
    return __builtin_shufflevector(lo, hi, 0,1,2,3,4,5,6,7,8,9,10,11,12,13,14,15);
}

// Cooperative A-tile load: A is bf16 row-major [M][Astride], 16B-aligned runs.
// Thread t covers 32 contiguous bf16 bytes-wise as 2x16B. Async path streams
// straight into LDS (ASYNCcnt); fallback stages through VGPRs (DScnt).
__device__ inline void load_a_tile(bf16_t* a_lds, const bf16_t* __restrict__ A,
                                   int m0, int k0, int Astride, int tid) {
    int m  = tid >> 1;
    int kh = (tid & 1) * 16;
    const bf16_t* src = &A[(size_t)(m0 + m) * Astride + k0 + kh];
    bf16_t* dst = &a_lds[m * ASTR + kh];
#if HAVE_ASYNC_LDS
    __builtin_amdgcn_global_load_async_to_lds_b128(
        (AS1G v4i*)src, (AS3L v4i*)dst, 0, 0);
    __builtin_amdgcn_global_load_async_to_lds_b128(
        (AS1G v4i*)(src + 8), (AS3L v4i*)(dst + 8), 0, 0);
#else
    v8bf lo = *(const v8bf*)src;
    v8bf hi = *(const v8bf*)(src + 8);
    *(v8bf*)dst       = lo;
    *(v8bf*)(dst + 8) = hi;
#endif
}

// ---------------- B-tile loader functors -----------------------------------
// Build B^T tile [TN][TK] in LDS: thread t handles n = t&63, k-run = (t>>6)*8.

// Implicit-GEMM conv B: B[k, n] = in[ci, n + kk] with k = ci*4 + kk.
// Branch once per 4-contiguous-element group; checks are compile-time where
// possible (only conv1 has KPAD != CIN*4).
template <typename TIN, int CIN, int LIN, int LCONV, int KPAD>
struct ConvB {
    const TIN* in;      // this batch's input: [CIN][LIN]
    __device__ void load(bf16_t* b_lds, int n0, int k0, int tid) const {
        int nl = tid & 63;
        int kh = (tid >> 6) * 8;
        int n  = n0 + nl;
        const bool nok = (n < LCONV);
        v8bf t;
#pragma unroll
        for (int g = 0; g < 2; ++g) {
            const int kb = k0 + kh + g * 4;   // multiple of 4
            const int ci = kb >> 2;
            bool ok = nok;
            if (CIN * 4 != KPAD) ok = ok && (ci < CIN);
            if (ok) {
                const TIN* p = &in[(size_t)ci * LIN + n];
                t[g * 4 + 0] = (bf16_t)(float)p[0];
                t[g * 4 + 1] = (bf16_t)(float)p[1];
                t[g * 4 + 2] = (bf16_t)(float)p[2];
                t[g * 4 + 3] = (bf16_t)(float)p[3];
            } else {
                t[g * 4 + 0] = (bf16_t)0.f;
                t[g * 4 + 1] = (bf16_t)0.f;
                t[g * 4 + 2] = (bf16_t)0.f;
                t[g * 4 + 3] = (bf16_t)0.f;
            }
        }
        *(v8bf*)&b_lds[nl * BSTR + kh] = t;
    }
};

// RNN scan B: rows 0..255 = X_t, rows 256..511 = H_prev. The X/H select
// depends only on k0 (uniform per K-step since 256 % TK == 0).
struct RnnB {
    const bf16_t* X;    // [256][BATCH]
    const bf16_t* H;    // [256][BATCH]
    __device__ void load(bf16_t* b_lds, int n0, int k0, int tid) const {
        int nl = tid & 63;
        int kh = (tid >> 6) * 8;
        int n  = n0 + nl;
        const bf16_t* S = (k0 < 256) ? (X + (size_t)k0 * BATCH)
                                     : (H + (size_t)(k0 - 256) * BATCH);
        v8bf t;
#pragma unroll
        for (int i = 0; i < 8; ++i)
            t[i] = S[(size_t)(kh + i) * BATCH + n];
        *(v8bf*)&b_lds[nl * BSTR + kh] = t;
    }
};

// RNN priming step: B := ones (only the X half is ever iterated, Kext=256).
struct PrimeB {
    __device__ void load(bf16_t* b_lds, int n0, int k0, int tid) const {
        int nl = tid & 63;
        int kh = (tid >> 6) * 8;
        v8bf t;
#pragma unroll
        for (int i = 0; i < 8; ++i) t[i] = (bf16_t)1.0f;
        *(v8bf*)&b_lds[nl * BSTR + kh] = t;
    }
};

// Scaled fp32 source (hsum/8 for the final linear).
struct ScaledB {
    const float* S;     // [K][BATCH]
    float scale;
    __device__ void load(bf16_t* b_lds, int n0, int k0, int tid) const {
        int nl = tid & 63;
        int kh = (tid >> 6) * 8;
        int n  = n0 + nl;
        v8bf t;
#pragma unroll
        for (int i = 0; i < 8; ++i)
            t[i] = (bf16_t)(S[(size_t)(k0 + kh + i) * BATCH + n] * scale);
        *(v8bf*)&b_lds[nl * BSTR + kh] = t;
    }
};

// ---------------- Shared WMMA GEMM core (double-buffered LDS) ---------------
// a_lds: [2][TM*ASTR], b_lds: [2][TN*BSTR]. One barrier per K-step; next tile
// streams into the alternate buffer while WMMA consumes the current one.
template <typename BL>
__device__ inline void gemm_tile(const bf16_t* __restrict__ A, int Kext, int Astride,
                                 int m0, int n0, const BL& bl,
                                 v8f acc[2][2], bf16_t* a_lds, bf16_t* b_lds) {
    const int tid  = threadIdx.x;
    const int lane = tid & 31;
    const int wid  = tid >> 5;
    const int wm   = (wid & 3) * 32;
    const int wn   = (wid >> 2) * 32;

    for (int mi = 0; mi < 2; ++mi)
        for (int ni = 0; ni < 2; ++ni)
#pragma unroll
            for (int r = 0; r < 8; ++r) acc[mi][ni][r] = 0.f;

    // Prologue: stage tile 0 into buffer 0.
    load_a_tile(a_lds, A, m0, 0, Astride, tid);
    bl.load(b_lds, n0, 0, tid);

    int buf = 0;
    for (int k0 = 0; k0 < Kext; k0 += TK) {
        async_wait_all();        // this wave's async LDS writes are visible
        __syncthreads();         // everyone's tile (buf) is complete

        const int nb = buf ^ 1;
        if (k0 + TK < Kext) {    // stream next K-slice into the other buffer
            load_a_tile(a_lds + nb * (TM * ASTR), A, m0, k0 + TK, Astride, tid);
            bl.load(b_lds + nb * (TN * BSTR), n0, k0 + TK, tid);
        }

        const bf16_t* al = a_lds + buf * (TM * ASTR);
        const bf16_t* bb = b_lds + buf * (TN * BSTR);
        v16bf af0 = load_a_frag(al, wm,      lane);
        v16bf af1 = load_a_frag(al, wm + 16, lane);
        v16bf bf0 = load_b_frag(bb, wn,      lane);
        v16bf bf1 = load_b_frag(bb, wn + 16, lane);

        acc[0][0] = __builtin_amdgcn_wmma_f32_16x16x32_bf16(false, af0, false, bf0,
                        (short)0, acc[0][0], false, false);
        acc[0][1] = __builtin_amdgcn_wmma_f32_16x16x32_bf16(false, af0, false, bf1,
                        (short)0, acc[0][1], false, false);
        acc[1][0] = __builtin_amdgcn_wmma_f32_16x16x32_bf16(false, af1, false, bf0,
                        (short)0, acc[1][0], false, false);
        acc[1][1] = __builtin_amdgcn_wmma_f32_16x16x32_bf16(false, af1, false, bf1,
                        (short)0, acc[1][1], false, false);
        buf = nb;
    }
}

// C/D 16x16 f32 layout: lane L -> n = L&15, m = (L>>4)*8 + r (r = vgpr 0..7).

// ---------------- Kernels ---------------------------------------------------

// Weight pack: fp32 [M][K] -> bf16 dst row-stride dstStride, columns
// [colOff, colOff+fill), zero-padded past K.
__global__ __launch_bounds__(BLOCK)
void pack_bf16_kernel(const float* __restrict__ src, bf16_t* __restrict__ dst,
                      int M, int K, int dstStride, int colOff, int fill) {
    size_t idx = (size_t)blockIdx.x * blockDim.x + threadIdx.x;
    size_t total = (size_t)M * fill;
    if (idx >= total) return;
    int kp = (int)(idx % fill);
    int m  = (int)(idx / fill);
    float v = (kp < K) ? src[(size_t)m * K + kp] : 0.f;
    dst[(size_t)m * dstStride + colOff + kp] = (bf16_t)v;
}

// Conv (implicit GEMM) + bias + relu -> bf16 [B][Cout][Lconv]
template <typename TIN, int CIN, int LIN, int COUT, int LCONV, int KPAD>
__global__ __launch_bounds__(BLOCK)
void conv_gemm_kernel(const TIN* __restrict__ in, const bf16_t* __restrict__ Aw,
                      const float* __restrict__ bias, bf16_t* __restrict__ out) {
    __shared__ bf16_t a_lds[2 * TM * ASTR];
    __shared__ bf16_t b_lds[2 * TN * BSTR];
    const int m0 = blockIdx.x * TM;
    const int n0 = blockIdx.y * TN;
    const int b  = blockIdx.z;

    ConvB<TIN, CIN, LIN, LCONV, KPAD> bl{in + (size_t)b * CIN * LIN};
    v8f acc[2][2];
    gemm_tile(Aw, KPAD, KPAD, m0, n0, bl, acc, a_lds, b_lds);

    const int lane = threadIdx.x & 31;
    const int wid  = threadIdx.x >> 5;
    const int wm   = (wid & 3) * 32;
    const int wn   = (wid >> 2) * 32;
#pragma unroll
    for (int mi = 0; mi < 2; ++mi)
#pragma unroll
        for (int ni = 0; ni < 2; ++ni) {
            int n = n0 + wn + ni * 16 + (lane & 15);
            if (n >= LCONV) continue;
            int mbase = m0 + wm + mi * 16 + (lane >> 4) * 8;
#pragma unroll
            for (int r = 0; r < 8; ++r) {
                int m = mbase + r;
                float v = acc[mi][ni][r] + bias[m];
                v = v > 0.f ? v : 0.f;
                out[((size_t)b * COUT + m) * LCONV + n] = (bf16_t)v;
            }
        }
}

// Max-pool 1d, bf16 [B][C][LIN] -> [B][C][LOUT]
template <int C, int LIN, int LOUT, int PK, int PS>
__global__ __launch_bounds__(BLOCK)
void pool_kernel(const bf16_t* __restrict__ in, bf16_t* __restrict__ out) {
    size_t idx = (size_t)blockIdx.x * blockDim.x + threadIdx.x;
    size_t total = (size_t)BATCH * C * LOUT;
    if (idx >= total) return;
    int l = (int)(idx % LOUT);
    size_t bc = idx / LOUT;
    const bf16_t* p = &in[bc * LIN + (size_t)l * PS];
    float m = (float)p[0];
#pragma unroll
    for (int j = 1; j < PK; ++j) { float v = (float)p[j]; m = v > m ? v : m; }
    out[idx] = (bf16_t)m;
}

// conv3 pool (k=3,s=2) fused with transpose: [B][256][18] -> [8][256][BATCH]
__global__ __launch_bounds__(BLOCK)
void pool3t_kernel(const bf16_t* __restrict__ in, bf16_t* __restrict__ out) {
    size_t idx = (size_t)blockIdx.x * blockDim.x + threadIdx.x;
    size_t total = (size_t)8 * 256 * BATCH;
    if (idx >= total) return;
    int b = (int)(idx % BATCH);
    size_t t = idx / BATCH;
    int c = (int)(t % 256);
    int l = (int)(t / 256);
    const bf16_t* p = &in[((size_t)b * 256 + c) * 18 + l * 2];
    float m = (float)p[0];
    float v1 = (float)p[1]; m = v1 > m ? v1 : m;
    float v2 = (float)p[2]; m = v2 > m ? v2 : m;
    out[idx] = (bf16_t)m;   // idx == (l*256 + c)*BATCH + b
}

// One RNN step: H_new = relu([Wih|Whh] @ [X;H] + b_ih + b_hh), states [256][BATCH].
// PRIME: X := ones, H := 0 -> only the Wih half contributes (Kext = 256).
template <bool PRIME>
__global__ __launch_bounds__(BLOCK)
void rnn_step_kernel(const bf16_t* __restrict__ Acat, const bf16_t* __restrict__ X,
                     const bf16_t* __restrict__ Hprev, const float* __restrict__ b_ih,
                     const float* __restrict__ b_hh, bf16_t* __restrict__ Hnew,
                     float* __restrict__ hsum, int step) {
    __shared__ bf16_t a_lds[2 * TM * ASTR];
    __shared__ bf16_t b_lds[2 * TN * BSTR];
    const int m0 = blockIdx.x * TM;
    const int n0 = blockIdx.y * TN;

    v8f acc[2][2];
    if (PRIME) {
        PrimeB bl{};
        gemm_tile(Acat, 256, 512, m0, n0, bl, acc, a_lds, b_lds);
    } else {
        RnnB bl{X, Hprev};
        gemm_tile(Acat, 512, 512, m0, n0, bl, acc, a_lds, b_lds);
    }

    const int lane = threadIdx.x & 31;
    const int wid  = threadIdx.x >> 5;
    const int wm   = (wid & 3) * 32;
    const int wn   = (wid >> 2) * 32;
#pragma unroll
    for (int mi = 0; mi < 2; ++mi)
#pragma unroll
        for (int ni = 0; ni < 2; ++ni) {
            int n = n0 + wn + ni * 16 + (lane & 15);
            int mbase = m0 + wm + mi * 16 + (lane >> 4) * 8;
#pragma unroll
            for (int r = 0; r < 8; ++r) {
                int m = mbase + r;
                float v = acc[mi][ni][r] + b_ih[m] + b_hh[m];
                v = v > 0.f ? v : 0.f;
                size_t o = (size_t)m * BATCH + n;
                Hnew[o] = (bf16_t)v;
                if (!PRIME) {
                    if (step == 1) hsum[o] = v;     // first scan step initializes
                    else           hsum[o] += v;    // (graph-replay safe, no memset)
                }
            }
        }
}

// Final linear: out[b][e] = (hsum/8) dot w_lin[e] + b_lin[e], fp32 output.
__global__ __launch_bounds__(BLOCK)
void linear_kernel(const bf16_t* __restrict__ Aw, const float* __restrict__ hsum,
                   const float* __restrict__ b_lin, float* __restrict__ out) {
    __shared__ bf16_t a_lds[2 * TM * ASTR];
    __shared__ bf16_t b_lds[2 * TN * BSTR];
    const int m0 = blockIdx.x * TM;
    const int n0 = blockIdx.y * TN;

    ScaledB bl{hsum, 0.125f};
    v8f acc[2][2];
    gemm_tile(Aw, 256, 256, m0, n0, bl, acc, a_lds, b_lds);

    const int lane = threadIdx.x & 31;
    const int wid  = threadIdx.x >> 5;
    const int wm   = (wid & 3) * 32;
    const int wn   = (wid >> 2) * 32;
#pragma unroll
    for (int mi = 0; mi < 2; ++mi)
#pragma unroll
        for (int ni = 0; ni < 2; ++ni) {
            int n = n0 + wn + ni * 16 + (lane & 15);
            int mbase = m0 + wm + mi * 16 + (lane >> 4) * 8;
#pragma unroll
            for (int r = 0; r < 8; ++r) {
                int m = mbase + r;
                out[(size_t)n * 1024 + m] = acc[mi][ni][r] + b_lin[m];
            }
        }
}

// ---------------- Host-side orchestration ----------------------------------
static inline unsigned cdivu(size_t a, size_t b) { return (unsigned)((a + b - 1) / b); }

extern "C" void kernel_launch(void* const* d_in, const int* in_sizes, int n_in,
                              void* d_out, int out_size, void* d_ws, size_t ws_size,
                              hipStream_t stream) {
    const float* x     = (const float*)d_in[0];
    const float* w1    = (const float*)d_in[1];
    const float* b1    = (const float*)d_in[2];
    const float* w2    = (const float*)d_in[3];
    const float* b2    = (const float*)d_in[4];
    const float* w3    = (const float*)d_in[5];
    const float* b3    = (const float*)d_in[6];
    const float* w_ih  = (const float*)d_in[7];
    const float* b_ih  = (const float*)d_in[8];
    const float* w_hh  = (const float*)d_in[9];
    const float* b_hh  = (const float*)d_in[10];
    const float* w_lin = (const float*)d_in[11];
    const float* b_lin = (const float*)d_in[12];
    float* out = (float*)d_out;

    // Workspace carve-out (deterministic layout every call).
    char* ws = (char*)d_ws;
    size_t off = 0;
    auto carve = [&](size_t bytes) {
        void* p = ws + off;
        off += (bytes + 255) & ~(size_t)255;
        return p;
    };
    bf16_t* w1b  = (bf16_t*)carve((size_t)384 * 288 * 2);
    bf16_t* w2b  = (bf16_t*)carve((size_t)512 * 1536 * 2);
    bf16_t* w3b  = (bf16_t*)carve((size_t)256 * 2048 * 2);
    bf16_t* wRb  = (bf16_t*)carve((size_t)256 * 512 * 2);     // [w_ih | w_hh]
    bf16_t* wLb  = (bf16_t*)carve((size_t)1024 * 256 * 2);
    bf16_t* H0   = (bf16_t*)carve((size_t)256 * BATCH * 2);
    bf16_t* H1   = (bf16_t*)carve((size_t)256 * BATCH * 2);
    float*  hsum = (float*) carve((size_t)256 * BATCH * 4);
    bf16_t* p3   = (bf16_t*)carve((size_t)8 * 256 * BATCH * 2);
    // Region B: pooled activations (p1 then p2), sized for the largest (p1).
    bf16_t* RB   = (bf16_t*)carve((size_t)BATCH * 384 * 66 * 2);
    // Region A: conv outputs (c1 then c2 then c3), sized for the largest (c1).
    bf16_t* RA   = (bf16_t*)carve((size_t)BATCH * 384 * 198 * 2);

    // --- pack weights to bf16 (A-matrices, K padded to multiple of 32) -----
    pack_bf16_kernel<<<cdivu((size_t)384 * 288, BLOCK),  BLOCK, 0, stream>>>(w1,    w1b, 384, 284, 288, 0, 288);
    pack_bf16_kernel<<<cdivu((size_t)512 * 1536, BLOCK), BLOCK, 0, stream>>>(w2,    w2b, 512, 1536, 1536, 0, 1536);
    pack_bf16_kernel<<<cdivu((size_t)256 * 2048, BLOCK), BLOCK, 0, stream>>>(w3,    w3b, 256, 2048, 2048, 0, 2048);
    pack_bf16_kernel<<<cdivu((size_t)256 * 256, BLOCK),  BLOCK, 0, stream>>>(w_ih,  wRb, 256, 256, 512, 0, 256);
    pack_bf16_kernel<<<cdivu((size_t)256 * 256, BLOCK),  BLOCK, 0, stream>>>(w_hh,  wRb, 256, 256, 512, 256, 256);
    pack_bf16_kernel<<<cdivu((size_t)1024 * 256, BLOCK), BLOCK, 0, stream>>>(w_lin, wLb, 1024, 256, 256, 0, 256);

    // --- conv1: [B,71,201] -> c1 [B,384,198] -> p1 [B,384,66] --------------
    conv_gemm_kernel<float, 71, 201, 384, 198, 288>
        <<<dim3(384 / TM, (198 + TN - 1) / TN, BATCH), BLOCK, 0, stream>>>(x, w1b, b1, RA);
    pool_kernel<384, 198, 66, 3, 3>
        <<<cdivu((size_t)BATCH * 384 * 66, BLOCK), BLOCK, 0, stream>>>(RA, RB);

    // --- conv2: p1 -> c2 [B,512,63] -> p2 [B,512,21] ------------------------
    conv_gemm_kernel<bf16_t, 384, 66, 512, 63, 1536>
        <<<dim3(512 / TM, (63 + TN - 1) / TN, BATCH), BLOCK, 0, stream>>>(RB, w2b, b2, RA);
    pool_kernel<512, 63, 21, 3, 3>
        <<<cdivu((size_t)BATCH * 512 * 21, BLOCK), BLOCK, 0, stream>>>(RA, RB);

    // --- conv3: p2 -> c3 [B,256,18] -> p3 [8][256][B] (pool k3 s2 + transpose)
    conv_gemm_kernel<bf16_t, 512, 21, 256, 18, 2048>
        <<<dim3(256 / TM, 1, BATCH), BLOCK, 0, stream>>>(RB, w3b, b3, RA);
    pool3t_kernel<<<cdivu((size_t)8 * 256 * BATCH, BLOCK), BLOCK, 0, stream>>>(RA, p3);

    // --- RNN: priming step (ones) + 8 scan steps, hsum accumulated ----------
    bf16_t* Hprev = H0;
    bf16_t* Hcur  = H1;
    rnn_step_kernel<true><<<dim3(256 / TM, BATCH / TN), BLOCK, 0, stream>>>(
        wRb, p3, Hprev, b_ih, b_hh, Hcur, hsum, 0);
    { bf16_t* t = Hprev; Hprev = Hcur; Hcur = t; }
    for (int s = 1; s <= 8; ++s) {
        const bf16_t* X = p3 + (size_t)(s - 1) * 256 * BATCH;
        rnn_step_kernel<false><<<dim3(256 / TM, BATCH / TN), BLOCK, 0, stream>>>(
            wRb, X, Hprev, b_ih, b_hh, Hcur, hsum, s);
        bf16_t* t = Hprev; Hprev = Hcur; Hcur = t;
    }

    // --- final linear: (hsum/8) @ w_lin.T + b_lin -> d_out [B,1024] fp32 ----
    linear_kernel<<<dim3(1024 / TM, BATCH / TN), BLOCK, 0, stream>>>(wLb, hsum, b_lin, out);
}
